// GIN_53352083751497
// MI455X (gfx1250) — matrix operand order repaired and checked by reading, not compiled, not run on hardware
//
#include <hip/hip_runtime.h>
#include <hip/hip_bf16.h>

// ---------------- problem constants ----------------
#define NN      50000
#define EE      800000
#define HH      128
#define H2      256
#define ND      64
#define ED      16
#define LAYERS  3
#define EPSV    1e-5f

typedef __bf16 bf16x16 __attribute__((ext_vector_type(16)));
typedef float  f32x8   __attribute__((ext_vector_type(8)));
typedef int    i32x4   __attribute__((vector_size(16)));

// ---------------- WMMA helper ----------------
__device__ __forceinline__ f32x8 wmma_bf16(bf16x16 a, bf16x16 b, f32x8 c) {
    return __builtin_amdgcn_wmma_f32_16x16x32_bf16(
        /*neg_a=*/false, a, /*neg_b=*/false, b,
        /*c_mod=*/(short)0, c, /*reuse_a=*/false, /*reuse_b=*/false);
}

// ---------------- async global->LDS weight staging ----------------
#if defined(__gfx1250__) && __has_builtin(__builtin_amdgcn_global_load_async_to_lds_b128)
#define HAVE_ASYNC_LDS 1
#else
#define HAVE_ASYNC_LDS 0
#endif

// Issue cooperative copy of n_elems bf16 (multiple of 8) from global to LDS.
__device__ __forceinline__ void stage_issue(__bf16* lds, const __bf16* g, int n_elems,
                                            int tid, int nthr) {
#if HAVE_ASYNC_LDS
    int nchunk = n_elems >> 3;            // 16B per async op
    for (int i = tid; i < nchunk; i += nthr) {
        __builtin_amdgcn_global_load_async_to_lds_b128(
            (__attribute__((address_space(1))) i32x4*)(const_cast<__bf16*>(g) + (size_t)i * 8),
            (__attribute__((address_space(3))) i32x4*)(lds + (size_t)i * 8),
            0, 0);
    }
#else
    int nchunk = n_elems >> 3;
    for (int i = tid; i < nchunk; i += nthr) {
        ((uint4*)lds)[i] = ((const uint4*)g)[i];
    }
#endif
}

__device__ __forceinline__ void stage_finish() {
#if HAVE_ASYNC_LDS
#if __has_builtin(__builtin_amdgcn_s_wait_asynccnt)
    __builtin_amdgcn_s_wait_asynccnt(0);
#else
    asm volatile("s_wait_asynccnt 0" ::: "memory");
#endif
#endif
    __syncthreads();
}

// A-fragment: 16x32 bf16 tile from f32 row-major [rows, ldk], rows mbase..mbase+15,
// K block kb..kb+31. Layout: lanes 0-15 hold K 0..15, lanes 16-31 hold
// K 16..31; element e -> k = kb+16*half+e.
__device__ __forceinline__ bf16x16 afrag(const float* A, int ldk, int mbase, int lane, int kb) {
    int r = mbase + (lane & 15);
    const float* p = A + (size_t)r * ldk + kb + ((lane >> 4) << 4);
    bf16x16 a;
#pragma unroll
    for (int i = 0; i < 8; ++i) a[i] = (__bf16)p[i];
#pragma unroll
    for (int i = 0; i < 8; ++i) a[8 + i] = (__bf16)p[8 + i];
    return a;
}

// B-fragment: 32x16 bf16 tile of W^T, W row-major bf16 [OUT, ldk] (global or LDS).
// B[k][n] = W[nbase + n][kb + k]. Lane layout: col = lane&15; lanes 0-15 hold
// K 0..15, lanes 16-31 hold K 16..31 -> one contiguous 16-element (32B) read.
__device__ __forceinline__ bf16x16 bfrag(const __bf16* W, int ldk, int nbase, int lane, int kb) {
    int n  = nbase + (lane & 15);
    int kh = (lane >> 4) << 4;
    return *reinterpret_cast<const bf16x16*>(W + (size_t)n * ldk + kb + kh);
}

// Edge-GEMM A-fragment: K=16 zero-padded to 32. half0 lanes take k=0..15,
// half1 lanes (k>=16) are zero.
__device__ __forceinline__ bf16x16 afrag_edge(const float* rowp, int lane) {
    const float* p = rowp;
    bf16x16 a;
#pragma unroll
    for (int i = 0; i < 16; ++i) a[i] = (lane < 16) ? (__bf16)p[i] : (__bf16)0.0f;
    return a;
}

// Edge-GEMM B-fragment (from LDS): rows K=0..15 valid (lanes 0-15), K=16..31 zero.
__device__ __forceinline__ bf16x16 bfrag_edge(const __bf16* W, int nbase, int lane) {
    int n = nbase + (lane & 15);
    bf16x16 b = *reinterpret_cast<const bf16x16*>(W + (size_t)n * ED);
    bool act = lane < 16;
#pragma unroll
    for (int i = 0; i < 16; ++i) b[i] = act ? b[i] : (__bf16)0.0f;
    return b;
}

// Fused bias + LayerNorm(H=128) + ReLU on 8 accumulator tiles (full 16x128 rows
// held by one wave). Row stats = per-lane partial over 8 tiles, then 16-lane
// shfl_xor reduction within each half-wave (rows r / r+8).
__device__ __forceinline__ void ln_relu_inplace(f32x8 acc[8], const float* bias,
                                                const float* g, const float* beta, int lane) {
    int lc = lane & 15;
    float bi[8], gg[8], bb[8];
#pragma unroll
    for (int t = 0; t < 8; ++t) {
        int c = t * 16 + lc;
        bi[t] = bias[c]; gg[t] = g[c]; bb[t] = beta[c];
    }
#pragma unroll
    for (int r = 0; r < 8; ++r) {
        float s = 0.f, q = 0.f;
#pragma unroll
        for (int t = 0; t < 8; ++t) {
            float v = acc[t][r] + bi[t];
            acc[t][r] = v;
            s += v; q += v * v;
        }
#pragma unroll
        for (int m = 1; m < 16; m <<= 1) {
            s += __shfl_xor(s, m, 32);
            q += __shfl_xor(q, m, 32);
        }
        float mean = s * (1.f / HH);
        float var  = q * (1.f / HH) - mean * mean;
        float inv  = rsqrtf(var + EPSV);
#pragma unroll
        for (int t = 0; t < 8; ++t) {
            float v = (acc[t][r] - mean) * inv * gg[t] + bb[t];
            acc[t][r] = fmaxf(v, 0.f);
        }
    }
}

// ---------------- small utility kernels ----------------
__global__ void cvt_bf16_kernel(const float* __restrict__ s, __bf16* __restrict__ d, int n) {
    int i = blockIdx.x * blockDim.x + threadIdx.x;
    if (i < n) d[i] = (__bf16)s[i];
}

__global__ void zero_f32_kernel(float* __restrict__ p, int n) {
    int i = blockIdx.x * blockDim.x + threadIdx.x;
    if (i < n) p[i] = 0.f;
}

// ---------------- node embed: h = relu(LN(x @ node_W^T + b)) ----------------
__global__ void node_embed_kernel(const float* __restrict__ x, const __bf16* __restrict__ Wb,
                                  const float* __restrict__ bias, const float* __restrict__ g,
                                  const float* __restrict__ beta, float* __restrict__ h) {
    __shared__ __attribute__((aligned(32))) __bf16 lW[HH * ND];   // 16 KB
    int tid = threadIdx.x;
    stage_issue(lW, Wb, HH * ND, tid, blockDim.x);
    stage_finish();

    int lane = tid & 31;
    int wave = blockIdx.x * (blockDim.x >> 5) + (tid >> 5);
    int mb = wave * 16;
    if (mb >= NN) return;                       // wave-uniform, after barrier

    bf16x16 a0 = afrag(x, ND, mb, lane, 0);
    bf16x16 a1 = afrag(x, ND, mb, lane, 32);
    f32x8 acc[8];
#pragma unroll
    for (int t = 0; t < 8; ++t) {
        f32x8 c = {};
        c = wmma_bf16(a0, bfrag(lW, ND, t * 16, lane, 0), c);
        c = wmma_bf16(a1, bfrag(lW, ND, t * 16, lane, 32), c);
        acc[t] = c;
    }
    ln_relu_inplace(acc, bias, g, beta, lane);

    int half = lane >> 4, lc = lane & 15;
#pragma unroll
    for (int r = 0; r < 8; ++r) {
        float* hr = h + (size_t)(mb + half * 8 + r) * HH;
#pragma unroll
        for (int t = 0; t < 8; ++t) hr[t * 16 + lc] = acc[t][r];
    }
}

// ---------------- fused edge embed + message + scatter-add ----------------
// rows 0..E-1: real edges (attr = edge_attr[e]); rows E..E+N-1: self loops
// (attr = sl_attr[l], src=dst=e-E). ea = relu(LN(attr @ edge_W^T + b));
// aggr[dst] += h[src] + ea.
__global__ void edge_msg_kernel(const float* __restrict__ edge_attr, const int* __restrict__ ei,
                                const float* __restrict__ sl, const __bf16* __restrict__ Wb,
                                const float* __restrict__ bias, const float* __restrict__ g,
                                const float* __restrict__ beta, const float* __restrict__ h,
                                float* __restrict__ aggr) {
    __shared__ int s_src[128];
    __shared__ int s_dst[128];
    __shared__ __attribute__((aligned(32))) __bf16 lW[HH * ED];   // 4 KB
    int tid  = threadIdx.x;
    int base = blockIdx.x * 128;

    stage_issue(lW, Wb, HH * ED, tid, blockDim.x);
    if (tid < 128) {
        int e = base + tid;
        int s = 0, d = 0;
        if (e < EE)            { s = ei[e]; d = ei[EE + e]; }
        else if (e < EE + NN)  { s = e - EE; d = s; }
        s_src[tid] = s; s_dst[tid] = d;
    }
    stage_finish();

    int lane = tid & 31, wv = tid >> 5;
    int mb = base + wv * 16;
    if (mb >= EE + NN) return;                  // wave-uniform, after barrier

    int er = mb + (lane & 15);
    const float* arow = (er < EE) ? (edge_attr + (size_t)er * ED) : sl;
    // software prefetch of the attr rows this wave-slot will touch ~32 blocks ahead
    if (er + 4096 < EE) __builtin_prefetch(edge_attr + (size_t)(er + 4096) * ED, 0, 1);
    bf16x16 a = afrag_edge(arow, lane);

    f32x8 acc[8];
#pragma unroll
    for (int t = 0; t < 8; ++t) {
        f32x8 c = {};
        acc[t] = wmma_bf16(a, bfrag_edge(lW, t * 16, lane), c);
    }
    ln_relu_inplace(acc, bias, g, beta, lane);

    int half = lane >> 4, lc = lane & 15;
#pragma unroll
    for (int r = 0; r < 8; ++r) {
        int li = wv * 16 + half * 8 + r;
        int s = s_src[li], d = s_dst[li];
        const float* hr = h + (size_t)s * HH;
        float* ar = aggr + (size_t)d * HH;
#pragma unroll
        for (int t = 0; t < 8; ++t) {
            int col = t * 16 + lc;
            atomicAdd(&ar[col], acc[t][r] + hr[col]);
        }
    }
}

// ---------------- MLP1: z = gelu(aggr @ W1^T + b1), out 256 cols ----------------
__global__ void mlp1_kernel(const float* __restrict__ aggr, const __bf16* __restrict__ W1,
                            const float* __restrict__ b1, float* __restrict__ z) {
    __shared__ __attribute__((aligned(32))) __bf16 lW[H2 * HH];   // 64 KB
    int tid = threadIdx.x;
    stage_issue(lW, W1, H2 * HH, tid, blockDim.x);
    stage_finish();

    int lane = tid & 31;
    int wave = blockIdx.x * (blockDim.x >> 5) + (tid >> 5);
    int mb = wave * 16;
    if (mb >= NN) return;

    bf16x16 a[4];
#pragma unroll
    for (int k = 0; k < 4; ++k) a[k] = afrag(aggr, HH, mb, lane, k * 32);

    int half = lane >> 4, lc = lane & 15;
#pragma unroll
    for (int t = 0; t < 16; ++t) {
        f32x8 c = {};
#pragma unroll
        for (int k = 0; k < 4; ++k)
            c = wmma_bf16(a[k], bfrag(lW, HH, t * 16, lane, k * 32), c);
        int col = t * 16 + lc;
        float bb = b1[col];
#pragma unroll
        for (int r = 0; r < 8; ++r) {
            float v = c[r] + bb;
            v = 0.5f * v * (1.f + erff(v * 0.70710678118654752f));  // exact gelu
            z[(size_t)(mb + half * 8 + r) * H2 + col] = v;
        }
    }
}

// ---------------- MLP2: hl = z @ W2^T + b2 ; accumulate BN col stats ----------------
__global__ void mlp2_kernel(const float* __restrict__ z, const __bf16* __restrict__ W2,
                            const float* __restrict__ b2, float* __restrict__ hout,
                            float* __restrict__ stats) {
    __shared__ __attribute__((aligned(32))) __bf16 lW[HH * H2];   // 64 KB
    int tid = threadIdx.x;
    stage_issue(lW, W2, HH * H2, tid, blockDim.x);
    stage_finish();

    int lane = tid & 31;
    int wave = blockIdx.x * (blockDim.x >> 5) + (tid >> 5);
    int mb = wave * 16;
    if (mb >= NN) return;

    bf16x16 a[8];
#pragma unroll
    for (int k = 0; k < 8; ++k) a[k] = afrag(z, H2, mb, lane, k * 32);

    int half = lane >> 4, lc = lane & 15;
#pragma unroll
    for (int t = 0; t < 8; ++t) {
        f32x8 c = {};
#pragma unroll
        for (int k = 0; k < 8; ++k)
            c = wmma_bf16(a[k], bfrag(lW, H2, t * 16, lane, k * 32), c);
        int col = t * 16 + lc;
        float bb = b2[col];
        float ps = 0.f, pq = 0.f;
#pragma unroll
        for (int r = 0; r < 8; ++r) {
            float v = c[r] + bb;
            hout[(size_t)(mb + half * 8 + r) * HH + col] = v;
            ps += v; pq += v * v;
        }
        // lanes l and l+16 hold the same column (rows 0-7 vs 8-15): combine first
        ps += __shfl_xor(ps, 16, 32);
        pq += __shfl_xor(pq, 16, 32);
        if (lane < 16) {
            atomicAdd(&stats[col], ps);
            atomicAdd(&stats[HH + col], pq);
        }
    }
}

// ---------------- BatchNorm apply (+optional ReLU) ----------------
__global__ void bn_apply_kernel(const float* __restrict__ hin, const float* __restrict__ stats,
                                const float* __restrict__ g, const float* __restrict__ b,
                                float* __restrict__ out, int relu) {
    int i = blockIdx.x * blockDim.x + threadIdx.x;
    if (i >= NN * HH) return;
    int c = i & (HH - 1);
    float mean = stats[c] * (1.f / NN);
    float var  = stats[HH + c] * (1.f / NN) - mean * mean;
    float v = (hin[i] - mean) * rsqrtf(var + EPSV) * g[c] + b[c];
    if (relu) v = fmaxf(v, 0.f);
    out[i] = v;
}

// ---------------- host launcher ----------------
extern "C" void kernel_launch(void* const* d_in, const int* in_sizes, int n_in,
                              void* d_out, int out_size, void* d_ws, size_t ws_size,
                              hipStream_t stream) {
    const float* x         = (const float*)d_in[0];
    const int*   ei        = (const int*)d_in[1];
    const float* edge_attr = (const float*)d_in[2];
    const float* node_W    = (const float*)d_in[3];
    const float* node_b    = (const float*)d_in[4];
    const float* node_ln_g = (const float*)d_in[5];
    const float* node_ln_b = (const float*)d_in[6];
    const float* edge_W    = (const float*)d_in[7];
    const float* edge_b    = (const float*)d_in[8];
    const float* edge_ln_g = (const float*)d_in[9];
    const float* edge_ln_b = (const float*)d_in[10];
    const float* sl_attr   = (const float*)d_in[11];
    const float* mlp_W1    = (const float*)d_in[12];
    const float* mlp_b1    = (const float*)d_in[13];
    const float* mlp_W2    = (const float*)d_in[14];
    const float* mlp_b2    = (const float*)d_in[15];
    const float* bn_g      = (const float*)d_in[16];
    const float* bn_b      = (const float*)d_in[17];
    float* out = (float*)d_out;

    // workspace layout (all 32B-aligned given 256B-aligned d_ws)
    float*  h_buf  = (float*)d_ws;                        // N*H
    float*  aggr   = h_buf + (size_t)NN * HH;             // N*H
    float*  z_buf  = aggr + (size_t)NN * HH;              // N*256
    float*  stats  = z_buf + (size_t)NN * H2;             // 256
    __bf16* nodeWb = (__bf16*)(stats + 256);              // 128*64
    __bf16* edgeWb = nodeWb + HH * ND;                    // 3*128*16
    __bf16* w1b    = edgeWb + LAYERS * HH * ED;           // 3*256*128
    __bf16* w2b    = w1b + (size_t)LAYERS * H2 * HH;      // 3*128*256

    const int NTHR = 256;
    const int NWAVE_NODE = NN / 16;                       // 3125
    const int NBLK_NODE  = (NWAVE_NODE + 7) / 8;          // 391
    const int NWAVE_EDGE = (EE + NN) / 16;                // 53125
    const int NBLK_EDGE  = (NWAVE_EDGE + 7) / 8;          // 6641

    // convert all weights to bf16 once
    cvt_bf16_kernel<<<(HH * ND + 255) / 256, 256, 0, stream>>>(node_W, nodeWb, HH * ND);
    cvt_bf16_kernel<<<(LAYERS * HH * ED + 255) / 256, 256, 0, stream>>>(edge_W, edgeWb, LAYERS * HH * ED);
    cvt_bf16_kernel<<<(LAYERS * H2 * HH + 255) / 256, 256, 0, stream>>>(mlp_W1, w1b, LAYERS * H2 * HH);
    cvt_bf16_kernel<<<(LAYERS * HH * H2 + 255) / 256, 256, 0, stream>>>(mlp_W2, w2b, LAYERS * HH * H2);

    node_embed_kernel<<<NBLK_NODE, NTHR, 0, stream>>>(x, nodeWb, node_b, node_ln_g, node_ln_b, h_buf);

    for (int l = 0; l < LAYERS; ++l) {
        zero_f32_kernel<<<(NN * HH + 255) / 256, 256, 0, stream>>>(aggr, NN * HH);
        edge_msg_kernel<<<NBLK_EDGE, NTHR, 0, stream>>>(
            edge_attr, ei, sl_attr + l * ED, edgeWb + (size_t)l * HH * ED,
            edge_b + l * HH, edge_ln_g + l * HH, edge_ln_b + l * HH, h_buf, aggr);
        mlp1_kernel<<<NBLK_NODE, NTHR, 0, stream>>>(
            aggr, w1b + (size_t)l * H2 * HH, mlp_b1 + l * H2, z_buf);
        zero_f32_kernel<<<1, 256, 0, stream>>>(stats, 256);
        mlp2_kernel<<<NBLK_NODE, NTHR, 0, stream>>>(
            z_buf, w2b + (size_t)l * HH * H2, mlp_b2 + l * HH, h_buf, stats);
        bn_apply_kernel<<<(NN * HH + 255) / 256, 256, 0, stream>>>(
            h_buf, stats, bn_g + l * HH, bn_b + l * HH,
            (l == LAYERS - 1) ? out : h_buf, (l != LAYERS - 1) ? 1 : 0);
    }
    (void)in_sizes; (void)n_in; (void)out_size; (void)ws_size;
}
